// GQAAttention_67482526154883
// MI455X (gfx1250) — matrix operand order repaired and checked by reading, not compiled, not run on hardware
//
#include <hip/hip_runtime.h>
#include <cstdint>

// ---------------------------------------------------------------- constants
static constexpr int B_   = 2;
static constexpr int S_   = 2048;
static constexpr int H_   = 2048;
static constexpr int NH_  = 16;
static constexpr int NKV_ = 4;
static constexpr int HD_  = 128;
static constexpr int G_   = NH_ / NKV_;       // 4
static constexpr int BS_  = B_ * S_;          // 4096
static constexpr float SCALE_ = 0.08838834764831845f; // 128^-0.5

// ---------------------------------------------------------------- types
typedef __attribute__((ext_vector_type(8)))  __bf16 v8bf;
typedef __attribute__((ext_vector_type(16))) __bf16 v16bf;
typedef __attribute__((ext_vector_type(8)))  float  v8f;
typedef __attribute__((ext_vector_type(4)))  unsigned int v4u;
typedef __attribute__((ext_vector_type(8)))  int v8i;
typedef __attribute__((ext_vector_type(4)))  int v4i;

// ---------------------------------------------------------------- WMMA helpers
__device__ __forceinline__ v8f wmma_bf16(v16bf a, v16bf b, v8f c) {
  // D = A(16x32) * B(32x16) + C, fp32 accumulate
  return __builtin_amdgcn_wmma_f32_16x16x32_bf16(
      /*neg_a=*/false, a, /*neg_b=*/false, b,
      /*c_mod=*/(short)0, c, /*reuse_a=*/false, /*reuse_b=*/false);
}

// A fragment: 16x32 bf16, source row-major [16][ld] (ld in elements).
// Lane l (l<16): row l, K {0..7}++{16..23}; lane l+16: row l, K {8..15}++{24..31}.
__device__ __forceinline__ v16bf frag_a(const __bf16* base, int ld, int lane) {
  int row = lane & 15;
  int ko  = (lane >> 4) * 8;
  const __bf16* p = base + row * ld + ko;
  v8bf lo = *(const v8bf*)(p);
  v8bf hi = *(const v8bf*)(p + 16);
  v16bf f;
#pragma unroll
  for (int i = 0; i < 8; ++i) { f[i] = lo[i]; f[8 + i] = hi[i]; }
  return f;
}

// B fragment: 32x16 of B sourced from B^T stored row-major [16 cols][ld>=32 K].
// Lane n (n<16): column n, K {0..7}++{16..23}; lane n+16: column n, K {8..15}++{24..31}.
__device__ __forceinline__ v16bf frag_b(const __bf16* base, int ld, int lane) {
  int col = lane & 15;
  int ko  = (lane >> 4) * 8;
  const __bf16* p = base + col * ld + ko;
  v8bf lo = *(const v8bf*)(p);
  v8bf hi = *(const v8bf*)(p + 16);
  v16bf f;
#pragma unroll
  for (int i = 0; i < 8; ++i) { f[i] = lo[i]; f[8 + i] = hi[i]; }
  return f;
}

// ---------------------------------------------------------------- TDM 2D tile load
// Build D# group0/group1 per CDNA5 ISA (CDNA5 ISA) and issue
// TENSOR_LOAD_TO_LDS. Dims/strides in elements (data_size = 2 bytes, bf16).
__device__ __forceinline__ void tdm_load_2d(unsigned lds_off, const void* gptr,
                                            unsigned tensor_d0, unsigned tensor_d1,
                                            unsigned long long stride0,
                                            unsigned tile_d0, unsigned tile_d1) {
  unsigned long long ga = (unsigned long long)(uintptr_t)gptr;
  v4u g0;
  g0[0] = 1u;                                              // count=1 (valid), user mode
  g0[1] = lds_off;                                         // lds_addr (bytes)
  g0[2] = (unsigned)ga;                                    // global_addr[31:0]
  g0[3] = (unsigned)((ga >> 32) & 0x1FFFFFFu) | (2u << 30);// addr[56:32] | type=2
  v8i g1;
  g1[0] = (int)(1u << 16);                                 // data_size=1 (2B), mask=0
  g1[1] = (int)((tensor_d0 & 0xFFFFu) << 16);              // tensor_dim0[15:0]
  g1[2] = (int)(((tensor_d0 >> 16) & 0xFFFFu) |
                ((tensor_d1 & 0xFFFFu) << 16));            // dim0[31:16] | dim1[15:0]
  g1[3] = (int)(((tensor_d1 >> 16) & 0xFFFFu) |
                ((tile_d0 & 0xFFFFu) << 16));              // dim1[31:16] | tile_dim0
  g1[4] = (int)(tile_d1 & 0xFFFFu);                        // tile_dim1 | tile_dim2=0
  g1[5] = (int)(unsigned)(stride0 & 0xFFFFFFFFull);        // tensor_dim0_stride[31:0]
  g1[6] = (int)(unsigned)((stride0 >> 32) & 0xFFFFull);    // stride[47:32] | stride1 lo
  g1[7] = 0;
  v4i z = {0, 0, 0, 0};
#if __clang_major__ >= 23
  v8i z8 = {0, 0, 0, 0, 0, 0, 0, 0};
  __builtin_amdgcn_tensor_load_to_lds(g0, g1, z, z, z8, 0);
#else
  __builtin_amdgcn_tensor_load_to_lds(g0, g1, z, z, 0);
#endif
}

// ---------------------------------------------------------------- fp32 -> bf16
__global__ void f32_to_bf16_kernel(const float* __restrict__ src,
                                   __bf16* __restrict__ dst, size_t n) {
  size_t i = (size_t)blockIdx.x * blockDim.x + threadIdx.x;
  size_t stride = (size_t)gridDim.x * blockDim.x;
  for (; i < n; i += stride) dst[i] = (__bf16)src[i];
}

// ---------------------------------------------------------------- bf16 GEMM
// C[M,N] (f32) = A[M,K]*B[K,N] + bias. M%128==0, N%128==0, K%32==0.
// 256 threads = 8 waves (4x2); block tile 128x128; wave tile 32x64 (8 WMMA/step).
// A tile staged by the Tensor Data Mover; B tile transposed into LDS manually.
__global__ __launch_bounds__(256) void gemm_bf16_kernel(
    const __bf16* __restrict__ A, const __bf16* __restrict__ Bm,
    const float* __restrict__ bias, float* __restrict__ C,
    int M, int N, int K) {
  __shared__ __bf16 sA[128 * 32];   // [row 128][k 32]
  __shared__ __bf16 sBT[128 * 32];  // [n 128][k 32]

  const int tid  = threadIdx.x;
  const int lane = tid & 31;
  const int wave = tid >> 5;        // 0..7
  const int wm   = wave >> 1;       // 0..3 -> 32 rows
  const int wn   = wave & 1;        // 0..1 -> 64 cols
  const int bm0  = blockIdx.y * 128;
  const int bn0  = blockIdx.x * 128;
  const unsigned sA_off = (unsigned)(uintptr_t)(void*)sA;

  v8f acc[2][4] = {};

  for (int k0 = 0; k0 < K; k0 += 32) {
    // --- A tile (128 rows x 32 k) via TDM async DMA, wave 0 issues
    if (tid < 32) {
      tdm_load_2d(sA_off, A + (size_t)bm0 * K + k0,
                  /*tensor_d0=*/32, /*tensor_d1=*/128,
                  /*stride0=*/(unsigned long long)K,
                  /*tile_d0=*/32, /*tile_d1=*/128);
    }
    // --- B tile transposed (32x128 -> sBT[128][32])
    {
      int kk = tid >> 3;            // 0..31
      int n0 = (tid & 7) * 16;      // 0..112
      const __bf16* g = Bm + (size_t)(k0 + kk) * N + bn0 + n0;
      v8bf v0 = *(const v8bf*)(g);
      v8bf v1 = *(const v8bf*)(g + 8);
#pragma unroll
      for (int i = 0; i < 8; ++i) {
        sBT[(n0 + i) * 32 + kk]     = v0[i];
        sBT[(n0 + 8 + i) * 32 + kk] = v1[i];
      }
    }
    if (tid < 32) __builtin_amdgcn_s_wait_tensorcnt(0);
    __syncthreads();

    v16bf af[2], bfr[4];
#pragma unroll
    for (int mi = 0; mi < 2; ++mi)
      af[mi] = frag_a(&sA[(wm * 32 + mi * 16) * 32], 32, lane);
#pragma unroll
    for (int ni = 0; ni < 4; ++ni)
      bfr[ni] = frag_b(&sBT[(wn * 64 + ni * 16) * 32], 32, lane);

#pragma unroll
    for (int mi = 0; mi < 2; ++mi)
#pragma unroll
      for (int ni = 0; ni < 4; ++ni)
        acc[mi][ni] = wmma_bf16(af[mi], bfr[ni], acc[mi][ni]);

    __syncthreads();
  }

  // --- epilogue: accum layout -> C. VGPR r: lanes 0-15 M=r, lanes 16-31 M=r+8.
  const int hi  = lane >> 4;
  const int col = lane & 15;
#pragma unroll
  for (int mi = 0; mi < 2; ++mi) {
#pragma unroll
    for (int ni = 0; ni < 4; ++ni) {
      int n = bn0 + wn * 64 + ni * 16 + col;
      float bval = bias ? bias[n] : 0.0f;
#pragma unroll
      for (int r = 0; r < 8; ++r) {
        int m = bm0 + wm * 32 + mi * 16 + hi * 8 + r;
        C[(size_t)m * N + n] = acc[mi][ni][r] + bval;
      }
    }
  }
}

// ---------------------------------------------------------------- RoPE + repack
// X: [B, S, nh, HD] f32 -> Out: [B, nh, S, HD] bf16 (rotated, optionally scaled)
__global__ void rope_pack_kernel(const float* __restrict__ X,
                                 const int* __restrict__ pos,
                                 __bf16* __restrict__ Out,
                                 int nh, float scale) {
  int t = blockIdx.x * blockDim.x + threadIdx.x;
  int total = B_ * S_ * nh * 64;
  if (t >= total) return;
  int d     = t & 63;
  int rest  = t >> 6;
  int hh    = rest % nh;
  int rest2 = rest / nh;
  int s     = rest2 % S_;
  int b     = rest2 / S_;

  float p    = (float)pos[b * S_ + s];
  float invf = __expf(-9.210340371976184f * (float)d * (1.0f / 64.0f)); // 10000^(-d/64)
  float ang  = p * invf;
  float sn, cs;
  sincosf(ang, &sn, &cs);

  const float* xp = X + (((size_t)b * S_ + s) * nh + hh) * HD_;
  float x1 = xp[d], x2 = xp[d + 64];
  __bf16* op = Out + (((size_t)b * nh + hh) * S_ + s) * HD_;
  op[d]      = (__bf16)((x1 * cs - x2 * sn) * scale);
  op[d + 64] = (__bf16)((x2 * cs + x1 * sn) * scale);
}

// V: [B, S, NKV, HD] f32 -> [B, NKV, S, HD] bf16
__global__ void pack_v_kernel(const float* __restrict__ V, __bf16* __restrict__ Out) {
  int t = blockIdx.x * blockDim.x + threadIdx.x;
  int total = B_ * S_ * NKV_ * HD_;
  if (t >= total) return;
  int d     = t & 127;
  int rest  = t >> 7;
  int kv    = rest % NKV_;
  int rest2 = rest / NKV_;
  int s     = rest2 % S_;
  int b     = rest2 / S_;
  Out[(((size_t)b * NKV_ + kv) * S_ + s) * HD_ + d] = (__bf16)V[t];
}

// ---------------------------------------------------------------- flash attention
// Qr:[B,NH,S,HD] bf16 (pre-scaled), Kr/Vr:[B,NKV,S,HD] bf16, Ao:[B,S,NH*HD] bf16
// 128 threads = 4 waves; wave owns 16 query rows; KV streamed 32 at a time.
// K tile staged by TDM; V tile transposed into LDS manually.
__global__ __launch_bounds__(128) void flash_attn_kernel(
    const __bf16* __restrict__ Qr, const __bf16* __restrict__ Kr,
    const __bf16* __restrict__ Vr, __bf16* __restrict__ Ao) {
  __shared__ __bf16 sK[32 * 128];       // K tile row-major: [c(32)][d(128)]
  __shared__ __bf16 sVT[128 * 32];      // V^T tile: [d(128)][c(32)]
  __shared__ __bf16 sP[4 * 16 * 32];    // per-wave P scratch (16x32)

  const int tid  = threadIdx.x;
  const int lane = tid & 31;
  const int wv   = tid >> 5;            // 0..3
  const int q0   = blockIdx.x * 64;
  const int h    = blockIdx.y;
  const int b    = blockIdx.z;
  const int kv   = h / G_;

  const __bf16* Qb = Qr + ((size_t)(b * NH_  + h)  * S_) * HD_;
  const __bf16* Kb = Kr + ((size_t)(b * NKV_ + kv) * S_) * HD_;
  const __bf16* Vb = Vr + ((size_t)(b * NKV_ + kv) * S_) * HD_;
  const int qt = q0 + wv * 16;
  const unsigned sK_off = (unsigned)(uintptr_t)(void*)sK;

  // preload Q fragments (16x128 = 4 fragments), already scaled by HD^-0.5
  v16bf qf[4];
#pragma unroll
  for (int dt = 0; dt < 4; ++dt)
    qf[dt] = frag_a(Qb + (size_t)qt * HD_ + dt * 32, HD_, lane);

  v8f o[8] = {};
  float mrun[8], lrun[8];
#pragma unroll
  for (int r = 0; r < 8; ++r) { mrun[r] = -3.0e38f; lrun[r] = 0.0f; }

  const int hi  = lane >> 4;
  const int col = lane & 15;
  const int cend = q0 + 64;             // causal upper bound for this block

  for (int c0 = 0; c0 < cend; c0 += 32) {
    // --- K tile (32 rows x 128) via TDM async DMA, wave 0 issues
    if (tid < 32) {
      tdm_load_2d(sK_off, Kb + (size_t)c0 * HD_,
                  /*tensor_d0=*/128, /*tensor_d1=*/32,
                  /*stride0=*/(unsigned long long)HD_,
                  /*tile_d0=*/128, /*tile_d1=*/32);
    }
    // --- stage V^T (32 rows of V -> sVT[d][c]) cooperatively
    {
      int c  = tid >> 2;                // 0..31
      int d0 = (tid & 3) * 32;          // 0,32,64,96
      const __bf16* g = Vb + (size_t)(c0 + c) * HD_ + d0;
#pragma unroll
      for (int j = 0; j < 4; ++j) {
        v8bf v = *(const v8bf*)(g + j * 8);
#pragma unroll
        for (int i = 0; i < 8; ++i) sVT[(d0 + j * 8 + i) * 32 + c] = v[i];
      }
    }
    // prefetch next V tile while computing this one
    if (c0 + 32 < cend)
      __builtin_prefetch(Vb + (size_t)(c0 + 32 + (lane & 31)) * HD_, 0, 3);
    if (tid < 32) __builtin_amdgcn_s_wait_tensorcnt(0);
    __syncthreads();

    // --- scores S = Q * K^T for two 16-column tiles (K rows in LDS)
    v8f s[2];
#pragma unroll
    for (int ct = 0; ct < 2; ++ct) {
      v8f acc = {};
#pragma unroll
      for (int dt = 0; dt < 4; ++dt) {
        v16bf bk = frag_b(&sK[(ct * 16) * 128 + dt * 32], 128, lane);
        acc = wmma_bf16(qf[dt], bk, acc);
      }
      // causal mask: this lane holds column c; rows qt + hi*8 + r
      int c = c0 + ct * 16 + col;
#pragma unroll
      for (int r = 0; r < 8; ++r) {
        int m = qt + hi * 8 + r;
        if (c > m) acc[r] = -1.0e9f;
      }
      s[ct] = acc;
    }

    // --- online softmax stats (row reductions across each 16-lane half)
    float alpha[8];
#pragma unroll
    for (int r = 0; r < 8; ++r) {
      float mx = fmaxf(s[0][r], s[1][r]);
      mx = fmaxf(mx, __shfl_xor(mx, 1, 32));
      mx = fmaxf(mx, __shfl_xor(mx, 2, 32));
      mx = fmaxf(mx, __shfl_xor(mx, 4, 32));
      mx = fmaxf(mx, __shfl_xor(mx, 8, 32));
      float mn = fmaxf(mrun[r], mx);
      alpha[r] = __expf(mrun[r] - mn);
      mrun[r]  = mn;
    }

    // --- P = exp(s - m); spill to per-wave LDS to relayout accum -> A fragment
    __bf16* Pw = &sP[wv * 16 * 32];
#pragma unroll
    for (int ct = 0; ct < 2; ++ct) {
#pragma unroll
      for (int r = 0; r < 8; ++r) {
        float p = __expf(s[ct][r] - mrun[r]);
        s[ct][r] = p;
        Pw[(hi * 8 + r) * 32 + ct * 16 + col] = (__bf16)p;
      }
    }
#pragma unroll
    for (int r = 0; r < 8; ++r) {
      float ps = s[0][r] + s[1][r];
      ps += __shfl_xor(ps, 1, 32);
      ps += __shfl_xor(ps, 2, 32);
      ps += __shfl_xor(ps, 4, 32);
      ps += __shfl_xor(ps, 8, 32);
      lrun[r] = lrun[r] * alpha[r] + ps;
    }

    // --- O = O*alpha + P * V   (8 d-tiles of 16)
    v16bf pf = frag_a(Pw, 32, lane);
#pragma unroll
    for (int dt = 0; dt < 8; ++dt) {
      v8f oo = o[dt];
#pragma unroll
      for (int r = 0; r < 8; ++r) oo[r] *= alpha[r];
      v16bf bv = frag_b(&sVT[(dt * 16) * 32], 32, lane);
      o[dt] = wmma_bf16(pf, bv, oo);
    }
    __syncthreads();
  }

  // --- epilogue: normalize and store bf16 head output
  float inv[8];
#pragma unroll
  for (int r = 0; r < 8; ++r) inv[r] = 1.0f / lrun[r];
#pragma unroll
  for (int dt = 0; dt < 8; ++dt) {
#pragma unroll
    for (int r = 0; r < 8; ++r) {
      int m = qt + hi * 8 + r;
      int d = dt * 16 + col;
      Ao[((size_t)b * S_ + m) * (NH_ * HD_) + h * HD_ + d] =
          (__bf16)(o[dt][r] * inv[r]);
    }
  }
}

// ---------------------------------------------------------------- launch
extern "C" void kernel_launch(void* const* d_in, const int* in_sizes, int n_in,
                              void* d_out, int out_size, void* d_ws, size_t ws_size,
                              hipStream_t stream) {
  const float* hidden = (const float*)d_in[0];
  const int*   pos    = (const int*)d_in[1];
  // d_in[2] attention_mask: pure causal, handled analytically
  const float* Wq = (const float*)d_in[3];
  const float* bq = (const float*)d_in[4];
  const float* Wk = (const float*)d_in[5];
  const float* bk = (const float*)d_in[6];
  const float* Wv = (const float*)d_in[7];
  const float* bv = (const float*)d_in[8];
  const float* Wo = (const float*)d_in[9];
  float* out = (float*)d_out;

  char* ws = (char*)d_ws;
  size_t off = 0;
  auto take = [&](size_t bytes) -> char* {
    char* p = ws + off;
    off += (bytes + 255) & ~(size_t)255;
    return p;
  };

  __bf16* hB  = (__bf16*)take((size_t)BS_ * H_ * 2);
  __bf16* WqB = (__bf16*)take((size_t)H_ * (NH_ * HD_) * 2);
  __bf16* WkB = (__bf16*)take((size_t)H_ * (NKV_ * HD_) * 2);
  __bf16* WvB = (__bf16*)take((size_t)H_ * (NKV_ * HD_) * 2);
  __bf16* WoB = (__bf16*)take((size_t)(NH_ * HD_) * H_ * 2);
  float*  Qf  = (float*)take((size_t)BS_ * (NH_ * HD_) * 4);
  float*  Kf  = (float*)take((size_t)BS_ * (NKV_ * HD_) * 4);
  float*  Vf  = (float*)take((size_t)BS_ * (NKV_ * HD_) * 4);
  __bf16* Qr  = (__bf16*)take((size_t)B_ * NH_  * S_ * HD_ * 2);
  __bf16* Kr  = (__bf16*)take((size_t)B_ * NKV_ * S_ * HD_ * 2);
  __bf16* Vr  = (__bf16*)take((size_t)B_ * NKV_ * S_ * HD_ * 2);
  __bf16* AoB = (__bf16*)take((size_t)BS_ * (NH_ * HD_) * 2);

  // 1) fp32 -> bf16 conversions
  auto cvt = [&](const float* s, __bf16* d, size_t n) {
    int blocks = (int)((n + 256 * 8 - 1) / (256 * 8));
    f32_to_bf16_kernel<<<blocks, 256, 0, stream>>>(s, d, n);
  };
  cvt(hidden, hB,  (size_t)BS_ * H_);
  cvt(Wq,     WqB, (size_t)H_ * NH_ * HD_);
  cvt(Wk,     WkB, (size_t)H_ * NKV_ * HD_);
  cvt(Wv,     WvB, (size_t)H_ * NKV_ * HD_);
  cvt(Wo,     WoB, (size_t)NH_ * HD_ * H_);

  // 2) QKV projections (WMMA GEMMs, TDM-staged A tiles)
  {
    dim3 gq(NH_ * HD_ / 128, BS_ / 128);
    gemm_bf16_kernel<<<gq, 256, 0, stream>>>(hB, WqB, bq, Qf, BS_, NH_ * HD_, H_);
    dim3 gk(NKV_ * HD_ / 128, BS_ / 128);
    gemm_bf16_kernel<<<gk, 256, 0, stream>>>(hB, WkB, bk, Kf, BS_, NKV_ * HD_, H_);
    gemm_bf16_kernel<<<gk, 256, 0, stream>>>(hB, WvB, bv, Vf, BS_, NKV_ * HD_, H_);
  }

  // 3) RoPE + head-major repack (fold 1/sqrt(HD) into Q)
  {
    int nq = B_ * S_ * NH_ * 64;
    rope_pack_kernel<<<(nq + 255) / 256, 256, 0, stream>>>(Qf, pos, Qr, NH_, SCALE_);
    int nk = B_ * S_ * NKV_ * 64;
    rope_pack_kernel<<<(nk + 255) / 256, 256, 0, stream>>>(Kf, pos, Kr, NKV_, 1.0f);
    int nv = B_ * S_ * NKV_ * HD_;
    pack_v_kernel<<<(nv + 255) / 256, 256, 0, stream>>>(Vf, Vr);
  }

  // 4) Flash attention (WMMA QK^T and PV, TDM-staged K tiles)
  {
    dim3 grid(S_ / 64, NH_, B_);
    flash_attn_kernel<<<grid, 128, 0, stream>>>(Qr, Kr, Vr, AoB);
  }

  // 5) Output projection -> fp32 d_out
  {
    dim3 go(H_ / 128, BS_ / 128);
    gemm_bf16_kernel<<<go, 256, 0, stream>>>(AoB, WoB, nullptr, out, BS_, H_, NH_ * HD_);
  }
}
